// NeuralEncoder_80152679678612
// MI455X (gfx1250) — matrix operand order repaired and checked by reading, not compiled, not run on hardware
//
#include <hip/hip_runtime.h>
#include <hip/hip_bf16.h>
#include <math.h>

// ---------------------------------------------------------------------------
// Dims (match reference)
// ---------------------------------------------------------------------------
constexpr int cB = 4, cT = 1024, cC = 256, cD = 256, cH = 512;
constexpr int cNH = 8, cHD = 64, cINTER = 2048, cL = 4;
constexpr int cCB = 128;               // context_backward (CF = 0)
constexpr int cM = cB * cT;            // 4096 rows

#define DEV __device__ __forceinline__

typedef __attribute__((ext_vector_type(16))) __bf16 bf16x16;
typedef __attribute__((ext_vector_type(4)))  __bf16 bf16x4;
typedef __attribute__((ext_vector_type(2)))  __bf16 bf16x2;
typedef __attribute__((ext_vector_type(8)))  float  f32x8;

union BF16x16 {
  bf16x16 v;
  bf16x2  h[8];
};

DEV bf16x2 pk2(float a, float b) {        // native v_cvt_pk_bf16_f32
  bf16x2 t; t.x = (__bf16)a; t.y = (__bf16)b; return t;
}

DEV float gelu_exact(float x) {
  return 0.5f * x * (1.0f + erff(x * 0.70710678118654752f));
}

// ds_swizzle_b32 xor-butterfly (group-of-32 mode: and=0x1f, or=0, xor=OFF)
template <int OFF>
DEV float swz_xor(float v) {
  return __int_as_float(
      __builtin_amdgcn_ds_swizzle(__float_as_int(v), (OFF << 10) | 0x1f));
}
DEV float redmax16(float v) {            // max across 16-lane half
  v = fmaxf(v, swz_xor<1>(v));
  v = fmaxf(v, swz_xor<2>(v));
  v = fmaxf(v, swz_xor<4>(v));
  v = fmaxf(v, swz_xor<8>(v));
  return v;
}
DEV float redsum32(float v) {            // sum across full wave32
  v += swz_xor<1>(v);  v += swz_xor<2>(v);
  v += swz_xor<4>(v);  v += swz_xor<8>(v);
  v += swz_xor<16>(v);
  return v;
}

// ---------------------------------------------------------------------------
// GEMM: C[M,N] = epi(A[M,K] @ W[K,N] + bias)   (f32 mem, bf16 WMMA, f32 acc)
// Block: 256 thr = 8 waves, tile 128x128, K-step 64 (2 WMMA k-chunks).
// Wave grid 4x2; each wave owns 32x64 = 2x4 tiles of 16x16.
// A staged row-major [m][k]; W staged TRANSPOSED [n][k] so both A and B
// fragments load contiguous bf16x2 pairs from LDS.
// ACT: 0 none, 1 exact-gelu.  RES: add resid[row*N+col] (resid may == C).
// ---------------------------------------------------------------------------
template <int ACT, int RES>
__global__ __launch_bounds__(256)
void gemm_kernel(const float* __restrict__ A, const float* __restrict__ W,
                 const float* __restrict__ bias, const float* __restrict__ resid,
                 float* __restrict__ C, int M, int N, int K) {
  constexpr int BK  = 64;
  constexpr int LDA = 72;    // bf16 elems per A LDS row (64 + pad 8)
  constexpr int LDW = 72;    // bf16 elems per Wt LDS row (64 + pad 8)
  __shared__ __bf16 As[128 * LDA];   // [m][k]
  __shared__ __bf16 Wt[128 * LDW];   // [n][k]  (transposed)

  const int tid  = threadIdx.x;
  const int wave = tid >> 5;
  const int lane = tid & 31;
  const int wm = wave >> 1;          // 0..3
  const int wn = wave & 1;           // 0..1
  const int m0 = blockIdx.y * 128;
  const int n0 = blockIdx.x * 128;

  f32x8 acc[2][4] = {};

  const int ml   = lane & 15;
  const int hi   = lane >> 4;        // lane half
  const int kb8  = hi * 8;           // unused; A frags also use kb16
  const int kb16 = hi * 16;          // B-frag K base per half

  for (int k0 = 0; k0 < K; k0 += BK) {
    // ---- stage A tile 128x64 (f32 -> bf16, row-major) ----
#pragma unroll
    for (int i = 0; i < 8; ++i) {
      int lin = i * 256 + tid;
      int row = lin >> 4, c4 = (lin & 15) * 4;
      const float4 av = *(const float4*)(A + (size_t)(m0 + row) * K + k0 + c4);
      bf16x4 t; t.x = (__bf16)av.x; t.y = (__bf16)av.y;
      t.z = (__bf16)av.z; t.w = (__bf16)av.w;
      *(bf16x4*)&As[row * LDA + c4] = t;
    }
    // ---- stage W tile 64x128 (f32 -> bf16, transposed into [n][k]) ----
#pragma unroll
    for (int i = 0; i < 8; ++i) {
      int lin = i * 256 + tid;
      int krow = lin >> 5, c4 = (lin & 31) * 4;
      const float4 wv = *(const float4*)(W + (size_t)(k0 + krow) * N + n0 + c4);
      Wt[(c4 + 0) * LDW + krow] = (__bf16)wv.x;
      Wt[(c4 + 1) * LDW + krow] = (__bf16)wv.y;
      Wt[(c4 + 2) * LDW + krow] = (__bf16)wv.z;
      Wt[(c4 + 3) * LDW + krow] = (__bf16)wv.w;
    }
    if (k0 + BK < K) {  // hint next tiles toward L2 (global_prefetch_b8)
      __builtin_prefetch(A + (size_t)(m0 + (tid >> 1)) * K + k0 + BK, 0, 1);
      __builtin_prefetch(W + (size_t)(k0 + BK + (tid >> 4)) * N + n0 + (tid & 15) * 8, 0, 1);
    }
    __syncthreads();

#pragma unroll
    for (int kc = 0; kc < 2; ++kc) {
      const int kof = kc * 32;
      // ---- A fragments (16x32 bf16: lane=M, K halves 0-15 / 16-31) ----
      BF16x16 af[2];
#pragma unroll
      for (int ms = 0; ms < 2; ++ms) {
        const __bf16* ap = &As[(wm * 32 + ms * 16 + ml) * LDA + kof + kb16];
#pragma unroll
        for (int vv = 0; vv < 4; ++vv) {
          af[ms].h[vv]     = *(const bf16x2*)(ap + 2 * vv);
          af[ms].h[vv + 4] = *(const bf16x2*)(ap + 8 + 2 * vv);
        }
      }
      // ---- B fragments (32x16 bf16: lane=N, K halves 0-15 / 16-31) ----
#pragma unroll
      for (int ns = 0; ns < 4; ++ns) {
        BF16x16 bfr;
        const __bf16* wp = &Wt[(wn * 64 + ns * 16 + ml) * LDW + kof + kb16];
#pragma unroll
        for (int vv = 0; vv < 8; ++vv) bfr.h[vv] = *(const bf16x2*)(wp + 2 * vv);
#pragma unroll
        for (int ms = 0; ms < 2; ++ms) {
          acc[ms][ns] = __builtin_amdgcn_wmma_f32_16x16x32_bf16(
              false, af[ms].v, false, bfr.v, (short)0, acc[ms][ns], false, false);
        }
      }
    }
    __syncthreads();
  }

  // ---- epilogue: bias (+gelu) (+resid) ----
#pragma unroll
  for (int ms = 0; ms < 2; ++ms) {
#pragma unroll
    for (int ns = 0; ns < 4; ++ns) {
#pragma unroll
      for (int r = 0; r < 8; ++r) {
        int row = m0 + wm * 32 + ms * 16 + r + hi * 8;
        int col = n0 + wn * 64 + ns * 16 + ml;
        float val = acc[ms][ns][r] + bias[col];
        if (ACT == 1) val = gelu_exact(val);
        if (RES) val += resid[(size_t)row * N + col];
        C[(size_t)row * N + col] = val;
      }
    }
  }
}

// ---------------------------------------------------------------------------
// LayerNorm over H=512, one wave per row (wave32: 16 elems/lane)
// ---------------------------------------------------------------------------
__global__ __launch_bounds__(256)
void ln_kernel(const float* __restrict__ x, const float* __restrict__ g,
               const float* __restrict__ b, float* __restrict__ y) {
  const int wave = threadIdx.x >> 5, lane = threadIdx.x & 31;
  const int row = blockIdx.x * 8 + wave;
  const float* xr = x + (size_t)row * cH;
  float vals[16], s = 0.f, s2 = 0.f;
#pragma unroll
  for (int i = 0; i < 16; ++i) {
    float t = xr[lane + i * 32];
    vals[i] = t; s += t; s2 += t * t;
  }
  s  = redsum32(s);
  s2 = redsum32(s2);
  const float mean = s * (1.0f / cH);
  const float var  = s2 * (1.0f / cH) - mean * mean;
  const float rinv = rsqrtf(var + 1e-5f);
  float* yr = y + (size_t)row * cH;
#pragma unroll
  for (int i = 0; i < 16; ++i) {
    int c = lane + i * 32;
    yr[c] = (vals[i] - mean) * rinv * g[c] + b[c];
  }
}

// ---------------------------------------------------------------------------
// RoPE in-place on q and k, one thread per (row, head, d<32) pair
// ---------------------------------------------------------------------------
__global__ __launch_bounds__(256)
void rope_kernel(float* __restrict__ q, float* __restrict__ k,
                 const int* __restrict__ ts) {
  int idx = blockIdx.x * blockDim.x + threadIdx.x;   // B*T*NH*32 threads
  int d   = idx & 31;
  int hh  = (idx >> 5) & 7;
  int row = idx >> 8;                 // 0..4095
  int bb  = row >> 10;
  int pos = row & 1023;
  float t   = (float)ts[bb * cT + pos];
  float inv = __expf(-(float)(2 * d) * (1.0f / cHD) * 9.210340371976184f); // ln(1e4)
  float ang = t * inv;
  float cs = cosf(ang), sn = sinf(ang);
  size_t base = (size_t)row * cH + hh * cHD + d;
  float q1 = q[base], q2 = q[base + 32];
  q[base]      = q1 * cs - q2 * sn;
  q[base + 32] = q2 * cs + q1 * sn;
  float k1 = k[base], k2 = k[base + 32];
  k[base]      = k1 * cs - k2 * sn;
  k[base + 32] = k2 * cs + k1 * sn;
}

// ---------------------------------------------------------------------------
// Banded flash attention. One wave per 16-row query tile.
// Window per row i: keys in [i-128, i]. Key tiles processed in pairs so PV is
// a K=32 bf16 WMMA; P relaid C->A layout via 1KB LDS bounce.
// Row sums of P computed by an extra WMMA against an all-ones B fragment
// (replaces a 4-step ds shuffle reduction per row per pair).
// 1/sqrt(HD) folded into the Q fragment. q,k,v,o layout: [B,T,NH,HD].
// ---------------------------------------------------------------------------
__global__ __launch_bounds__(32)
void attn_kernel(const float* __restrict__ q, const float* __restrict__ k,
                 const float* __restrict__ v, const int* __restrict__ smask,
                 float* __restrict__ o) {
  __shared__ __bf16 pbuf[16 * 32];   // P pair tile, row-major [m][key]

  const int lane = threadIdx.x;
  const int ml   = lane & 15;
  const int hi   = lane >> 4;
  const int kb8  = hi * 8;

  const int bb = blockIdx.z;
  const int hh = blockIdx.y;
  const int q0 = blockIdx.x * 16;

  // ---- preload Q A-fragments (two K=32 chunks of HD=64), pre-scaled ----
  const float* qp = q + ((size_t)(bb * cT + q0 + ml) * cH + hh * cHD);
  BF16x16 qa[2];
#pragma unroll
  for (int c = 0; c < 2; ++c) {
    const float4* a0 = (const float4*)(qp + c * 32 + hi * 16);
    const float4* a1 = (const float4*)(qp + c * 32 + hi * 16 + 8);
    float4 x0 = a0[0], x1 = a0[1], y0 = a1[0], y1 = a1[1];
    qa[c].h[0] = pk2(x0.x * 0.125f, x0.y * 0.125f);
    qa[c].h[1] = pk2(x0.z * 0.125f, x0.w * 0.125f);
    qa[c].h[2] = pk2(x1.x * 0.125f, x1.y * 0.125f);
    qa[c].h[3] = pk2(x1.z * 0.125f, x1.w * 0.125f);
    qa[c].h[4] = pk2(y0.x * 0.125f, y0.y * 0.125f);
    qa[c].h[5] = pk2(y0.z * 0.125f, y0.w * 0.125f);
    qa[c].h[6] = pk2(y1.x * 0.125f, y1.y * 0.125f);
    qa[c].h[7] = pk2(y1.z * 0.125f, y1.w * 0.125f);
  }

  // all-ones B fragment for row-sum WMMA
  BF16x16 ones;
#pragma unroll
  for (int vv = 0; vv < 8; ++vv) ones.h[vv] = pk2(1.0f, 1.0f);

  f32x8 acc[4] = {};
  f32x8 accl = {};                       // running row sums (every lane of a row)
  float mrow[8];
#pragma unroll
  for (int r = 0; r < 8; ++r) mrow[r] = -3.0e38f;

  const int kt_lo = (q0 >= cCB) ? ((q0 - cCB) >> 4) : 0;
  const int kt_hi = q0 >> 4;

  for (int t0 = kt_lo; t0 <= kt_hi; t0 += 2) {
    // ---- scores for the pair (tile t0+1 may be fully masked -> p==0) ----
    f32x8 s[2];
#pragma unroll
    for (int u = 0; u < 2; ++u) {
      const int kcol0 = (t0 + u) * 16;
      const int key   = kcol0 + ml;
      const int kcl   = key < (cT - 1) ? key : (cT - 1);
      f32x8 sc = {};
#pragma unroll
      for (int c = 0; c < 2; ++c) {
        BF16x16 kf;
        const float4* kp4 =
            (const float4*)(k + ((size_t)(bb * cT + kcl) * cH + hh * cHD + c * 32 + hi * 16));
        float4 a = kp4[0], b4 = kp4[1], c4 = kp4[2], d4 = kp4[3];
        kf.h[0] = pk2(a.x, a.y);  kf.h[1] = pk2(a.z, a.w);
        kf.h[2] = pk2(b4.x, b4.y); kf.h[3] = pk2(b4.z, b4.w);
        kf.h[4] = pk2(c4.x, c4.y); kf.h[5] = pk2(c4.z, c4.w);
        kf.h[6] = pk2(d4.x, d4.y); kf.h[7] = pk2(d4.z, d4.w);
        sc = __builtin_amdgcn_wmma_f32_16x16x32_bf16(
            false, qa[c].v, false, kf.v, (short)0, sc, false, false);
      }
      const int mok = smask[bb * cT + kcl];
#pragma unroll
      for (int r = 0; r < 8; ++r) {
        int row = q0 + r + hi * 8;
        bool ok = (key <= row) && (key >= row - cCB) && (mok > 0);
        s[u][r] = ok ? sc[r] : -3.0e38f;
      }
    }
    // ---- joint online softmax over the 32-key pair ----
    float mnew[8], alpha[8];
#pragma unroll
    for (int r = 0; r < 8; ++r) {
      float mx  = redmax16(fmaxf(s[0][r], s[1][r]));
      mnew[r]   = fmaxf(mrow[r], mx);
      alpha[r]  = __expf(mrow[r] - mnew[r]);
      mrow[r]   = mnew[r];
    }
#pragma unroll
    for (int r = 0; r < 8; ++r) {
      float pv0 = __expf(s[0][r] - mnew[r]);
      float pv1 = __expf(s[1][r] - mnew[r]);
      pbuf[(r + hi * 8) * 32 + ml]      = (__bf16)pv0;
      pbuf[(r + hi * 8) * 32 + 16 + ml] = (__bf16)pv1;
    }
    asm volatile("s_wait_dscnt 0" ::: "memory");
    // ---- rescale accumulators ----
#pragma unroll
    for (int r = 0; r < 8; ++r) {
      accl[r] *= alpha[r];
#pragma unroll
      for (int j = 0; j < 4; ++j) acc[j][r] *= alpha[r];
    }
    // ---- P A-fragment (16x32) from LDS ----
    BF16x16 pa;
#pragma unroll
    for (int vv = 0; vv < 4; ++vv) {
      pa.h[vv]     = *(const bf16x2*)&pbuf[ml * 32 + hi * 16 + 2 * vv];
      pa.h[vv + 4] = *(const bf16x2*)&pbuf[ml * 32 + 8 + hi * 16 + 2 * vv];
    }
    // ---- row sums: P x ones (all columns hold the row sum) ----
    accl = __builtin_amdgcn_wmma_f32_16x16x32_bf16(
        false, pa.v, false, ones.v, (short)0, accl, false, false);
    // ---- PV: V B-fragments 32x16 per HD chunk, key rows clamped ----
    const int keybase = t0 * 16 + hi * 16;
#pragma unroll
    for (int j = 0; j < 4; ++j) {
      BF16x16 vf;
#pragma unroll
      for (int vv = 0; vv < 8; ++vv) {
        int k0i = keybase + 2 * vv;
        int k0c = k0i < (cT - 1) ? k0i : (cT - 1);
        int k1c = (k0i + 1) < (cT - 1) ? (k0i + 1) : (cT - 1);
        vf.h[vv] = pk2(v[(size_t)(bb * cT + k0c) * cH + hh * cHD + j * 16 + ml],
                       v[(size_t)(bb * cT + k1c) * cH + hh * cHD + j * 16 + ml]);
      }
      acc[j] = __builtin_amdgcn_wmma_f32_16x16x32_bf16(
          false, pa.v, false, vf.v, (short)0, acc[j], false, false);
    }
  }

  // ---- normalize + store ----
#pragma unroll
  for (int j = 0; j < 4; ++j)
#pragma unroll
    for (int r = 0; r < 8; ++r) {
      int row = q0 + r + hi * 8;
      o[(size_t)(bb * cT + row) * cH + hh * cHD + j * 16 + ml] = acc[j][r] / accl[r];
    }
}

// ---------------------------------------------------------------------------
// Host-side orchestration
// ---------------------------------------------------------------------------
extern "C" void kernel_launch(void* const* d_in, const int* in_sizes, int n_in,
                              void* d_out, int out_size, void* d_ws, size_t ws_size,
                              hipStream_t stream) {
  const float* spikes  = (const float*)d_in[0];
  const int*   smask   = (const int*)  d_in[1];
  const int*   ts      = (const int*)  d_in[2];
  const float* embed_w = (const float*)d_in[3];
  const float* embed_b = (const float*)d_in[4];
  const float* proj_w  = (const float*)d_in[5];
  const float* proj_b  = (const float*)d_in[6];
  const float* ln1_g   = (const float*)d_in[7];
  const float* ln1_b   = (const float*)d_in[8];
  const float* Wq      = (const float*)d_in[9];
  const float* bq      = (const float*)d_in[10];
  const float* Wk      = (const float*)d_in[11];
  const float* bk      = (const float*)d_in[12];
  const float* Wv      = (const float*)d_in[13];
  const float* bv      = (const float*)d_in[14];
  const float* Wo      = (const float*)d_in[15];
  const float* bo      = (const float*)d_in[16];
  const float* ln2_g   = (const float*)d_in[17];
  const float* ln2_b   = (const float*)d_in[18];
  const float* up_w    = (const float*)d_in[19];
  const float* up_b    = (const float*)d_in[20];
  const float* down_w  = (const float*)d_in[21];
  const float* down_b  = (const float*)d_in[22];

  // workspace: x | h | q | k | v  (each M*H f32)  | mid (M*INTER f32)  ~72MB
  float* x   = (float*)d_ws;
  float* h   = x + (size_t)cM * cH;
  float* q   = h + (size_t)cM * cH;
  float* kq  = q + (size_t)cM * cH;
  float* vq  = kq + (size_t)cM * cH;
  float* mid = vq + (size_t)cM * cH;

  const dim3 blk(256);

  // embedding: gelu(spikes@embed_w+b) @ proj_w + b
  gemm_kernel<1, 0><<<dim3(cD / 128, cM / 128), blk, 0, stream>>>(
      spikes, embed_w, embed_b, nullptr, mid, cM, cD, cC);
  gemm_kernel<0, 0><<<dim3(cH / 128, cM / 128), blk, 0, stream>>>(
      mid, proj_w, proj_b, nullptr, x, cM, cH, cD);

  for (int l = 0; l < cL; ++l) {
    ln_kernel<<<cM / 8, blk, 0, stream>>>(x, ln1_g + l * cH, ln1_b + l * cH, h);
    gemm_kernel<0, 0><<<dim3(cH / 128, cM / 128), blk, 0, stream>>>(
        h, Wq + (size_t)l * cH * cH, bq + l * cH, nullptr, q, cM, cH, cH);
    gemm_kernel<0, 0><<<dim3(cH / 128, cM / 128), blk, 0, stream>>>(
        h, Wk + (size_t)l * cH * cH, bk + l * cH, nullptr, kq, cM, cH, cH);
    gemm_kernel<0, 0><<<dim3(cH / 128, cM / 128), blk, 0, stream>>>(
        h, Wv + (size_t)l * cH * cH, bv + l * cH, nullptr, vq, cM, cH, cH);
    rope_kernel<<<(cM * cNH * 32) / 256, blk, 0, stream>>>(q, kq, ts);
    attn_kernel<<<dim3(cT / 16, cNH, cB), dim3(32), 0, stream>>>(q, kq, vq, smask, h);
    gemm_kernel<0, 1><<<dim3(cH / 128, cM / 128), blk, 0, stream>>>(
        h, Wo + (size_t)l * cH * cH, bo + l * cH, x, x, cM, cH, cH);
    ln_kernel<<<cM / 8, blk, 0, stream>>>(x, ln2_g + l * cH, ln2_b + l * cH, h);
    gemm_kernel<1, 0><<<dim3(cINTER / 128, cM / 128), blk, 0, stream>>>(
        h, up_w + (size_t)l * cH * cINTER, up_b + l * cINTER, nullptr, mid, cM, cINTER, cH);
    gemm_kernel<0, 1><<<dim3(cH / 128, cM / 128), blk, 0, stream>>>(
        mid, down_w + (size_t)l * cINTER * cH, down_b + l * cH, x, x, cM, cH, cINTER);
  }

  hipMemcpyAsync(d_out, x, (size_t)out_size * sizeof(float),
                 hipMemcpyDeviceToDevice, stream);
}
